// QLSTMGen_65481071403478
// MI455X (gfx1250) — compile-verified, matrix operand order checked by reading, not run on hardware
//
#include <hip/hip_runtime.h>
#include <hip/hip_bf16.h>

// ---------------- types ----------------
typedef _Float16 h4  __attribute__((ext_vector_type(4)));
typedef _Float16 h8  __attribute__((ext_vector_type(8)));
typedef _Float16 v16h __attribute__((ext_vector_type(16)));
typedef float    v8f  __attribute__((ext_vector_type(8)));
typedef int      v4i  __attribute__((vector_size(16)));   // matches builtin proto

union F16x16 { v16h v; h8 h[2]; };

__device__ __forceinline__ v8f splat8(float b) {
  v8f v;
#pragma unroll
  for (int i = 0; i < 8; ++i) v[i] = b;
  return v;
}

// Load one A/B fragment for v_wmma_f32_16x16x32_f16.
// p0 points at (row = tile_row + l16)*ld + k0 + half*8 in a row-major f16 matrix.
__device__ __forceinline__ v16h frag_ld(const _Float16* p0) {
  F16x16 u;
  u.h[0] = *(const h8*)(p0);
  u.h[1] = *(const h8*)(p0 + 16);
  return u.v;
}

__device__ __forceinline__ v8f wmma16(v16h a, v16h b, v8f c) {
  return __builtin_amdgcn_wmma_f32_16x16x32_f16(false, a, false, b, (short)0, c,
                                                false, false);
}

// ---------------- async global->LDS (CDNA5) ----------------
#ifndef __has_builtin
#define __has_builtin(x) 0
#endif
#if __has_builtin(__builtin_amdgcn_global_load_async_to_lds_b128)
#define USE_ASYNC 1
#else
#define USE_ASYNC 0
#endif

__device__ __forceinline__ void cp_async16(void* ldst, const void* gsrc) {
#if USE_ASYNC
  __builtin_amdgcn_global_load_async_to_lds_b128(
      (__attribute__((address_space(1))) v4i*)gsrc,
      (__attribute__((address_space(3))) v4i*)ldst, 0, 0);
#else
  *(uint4*)ldst = *(const uint4*)gsrc;
#endif
}

__device__ __forceinline__ void async_wait0() {
#if USE_ASYNC
#if __has_builtin(__builtin_amdgcn_s_wait_asynccnt)
  __builtin_amdgcn_s_wait_asynccnt(0);
#else
  asm volatile("s_wait_asynccnt 0" ::: "memory");
#endif
#endif
}

__device__ __forceinline__ float sigm(float x) { return 1.f / (1.f + __expf(-x)); }

// ---------------- constants ----------------
// S=1024 B=64 D=128 H=256 K=D+H=384 NQ=8 NH=4 HD=64
#define COMBLD 392   // 64 rows x 392 halves (16B-aligned row stride)

// ---------------- f32 -> f16 weight convert ----------------
__global__ void cvt_f32_to_f16(const float* __restrict__ src,
                               _Float16* __restrict__ dst, int n) {
  int i = blockIdx.x * blockDim.x + threadIdx.x;
  if (i < n) dst[i] = (_Float16)src[i];
}

// ---------------- persistent LSTM scan (1 block, 1024 threads = 32 waves) ----
__global__ __launch_bounds__(1024, 1) void scan_kernel(
    const float* __restrict__ x,          // (S,B,D) f32
    const _Float16* __restrict__ Wc4,     // 4 x (256 x 384) f16 gate weights
    const _Float16* __restrict__ Wq,      // 32 x 384 f16 stacked quantum in-proj
    const float* __restrict__ bf, const float* __restrict__ bi,
    const float* __restrict__ bg, const float* __restrict__ bo,
    const float* __restrict__ bfq, const float* __restrict__ biq,
    const float* __restrict__ bgq, const float* __restrict__ boq,
    const float* __restrict__ Wqh,        // 256 x 8 f32
    const float* __restrict__ bqh,        // 256 f32
    const float* __restrict__ mixp,
    float* __restrict__ outs,             // (S,B,H) f32   -> d_out
    _Float16* __restrict__ outsH,         // (S,B,H) f16   -> ws
    float* __restrict__ hxo,              // (B,H) f32
    float* __restrict__ cxo)              // (B,H) f32
{
  __shared__ _Float16 comb[64 * COMBLD];  // [x(128) | hx(256)] as f16
  __shared__ float qpre[64 * 32];         // quantum pre-activations

  const int tid  = threadIdx.x;
  const int lane = tid & 31, wid = tid >> 5;
  const int l16  = lane & 15, half = lane >> 4;
  const int h8o  = half * 8;
  const float mixv = mixp[0];

  // two (16x16) output tiles per wave over the 64x256 hidden space:
  // tile tt: m0=(tt>>4)*16 rows of batch, j0=(tt&15)*16 cols of hidden.
  int m0s[2], j0s[2];
  m0s[0] = ((wid * 2) >> 4) * 16;     j0s[0] = ((wid * 2) & 15) * 16;
  m0s[1] = ((wid * 2 + 1) >> 4) * 16; j0s[1] = ((wid * 2 + 1) & 15) * 16;

  float bfv[2], biv[2], bgv[2], bov[2], bqhv[2], wqv[2][8];
#pragma unroll
  for (int a = 0; a < 2; ++a) {
    int j = j0s[a] + l16;
    bfv[a] = bf[j]; biv[a] = bi[j]; bgv[a] = bg[j]; bov[a] = bo[j];
    bqhv[a] = bqh[j];
#pragma unroll
    for (int p = 0; p < 8; ++p) wqv[a][p] = Wqh[j * 8 + p];
  }

  // qpre tile assignment: waves 0..7 each own one 16x16 tile of the 64x32 qpre.
  const int qm0 = (wid >> 1) * 16, qn0 = (wid & 1) * 16;
  float bq0 = 0.f;
  if (wid < 8) {
    int n = qn0 + l16;  // 0..31 -> gate = n>>3, idx = n&7
    const float* bql[4] = {bfq, biq, bgq, boq};
    bq0 = bql[n >> 3][n & 7];
  }

  // zero hx region of comb
  for (int i = tid; i < 64 * 256; i += 1024) {
    int m = i >> 8, j = i & 255;
    comb[m * COMBLD + 128 + j] = (_Float16)0.f;
  }

  float cxv[2][8];
#pragma unroll
  for (int a = 0; a < 2; ++a)
#pragma unroll
    for (int e = 0; e < 8; ++e) cxv[a][e] = 0.f;

  for (int t = 0; t < 1024; ++t) {
    // ---- phase A: convert x[t] (f32) into comb[:,0:128) as f16 ----
    const float4* xt = (const float4*)(x + (size_t)t * 64 * 128);
#pragma unroll
    for (int i = 0; i < 2; ++i) {
      int q = tid + i * 1024;       // 2048 float4 chunks
      int m = q >> 5, c = (q & 31) * 4;
      float4 v = xt[q];
      h4 hv; hv[0] = (_Float16)v.x; hv[1] = (_Float16)v.y;
      hv[2] = (_Float16)v.z; hv[3] = (_Float16)v.w;
      *(h4*)(comb + m * COMBLD + c) = hv;
    }
    __syncthreads();

    // ---- phase B: quantum pre-proj (waves 0-7) + classical gate WMMA (all) ----
    if (wid < 8) {
      v8f acc = splat8(bq0);
#pragma unroll
      for (int ks = 0; ks < 12; ++ks) {
        int k0 = ks * 32;
        v16h A = frag_ld(comb + (qm0 + l16) * COMBLD + k0 + h8o);
        v16h B = frag_ld(Wq + (qn0 + l16) * 384 + k0 + h8o);
        acc = wmma16(A, B, acc);
      }
#pragma unroll
      for (int e = 0; e < 8; ++e)
        qpre[(qm0 + e + 8 * half) * 32 + qn0 + l16] = acc[e];
    }

    v8f af[2], ai_[2], ag[2], ao[2];
#pragma unroll
    for (int a = 0; a < 2; ++a) {
      af[a] = splat8(bfv[a]); ai_[a] = splat8(biv[a]);
      ag[a] = splat8(bgv[a]); ao[a] = splat8(bov[a]);
      const int m0 = m0s[a], j0 = j0s[a];
      const _Float16* wrow = Wc4 + (size_t)(j0 + l16) * 384;
#pragma unroll
      for (int ks = 0; ks < 12; ++ks) {
        int k0 = ks * 32;
        v16h A = frag_ld(comb + (m0 + l16) * COMBLD + k0 + h8o);
        v16h Bf = frag_ld(wrow + (size_t)0 * 98304 + k0 + h8o);
        af[a] = wmma16(A, Bf, af[a]);
        v16h Bi = frag_ld(wrow + (size_t)1 * 98304 + k0 + h8o);
        ai_[a] = wmma16(A, Bi, ai_[a]);
        v16h Bg = frag_ld(wrow + (size_t)2 * 98304 + k0 + h8o);
        ag[a] = wmma16(A, Bg, ag[a]);
        v16h Bo = frag_ld(wrow + (size_t)3 * 98304 + k0 + h8o);
        ao[a] = wmma16(A, Bo, ao[a]);
      }
    }
    __syncthreads();  // qpre visible; all comb reads done

    // ---- phase C: mix, cell update (cx in registers), write hx ----
#pragma unroll
    for (int a = 0; a < 2; ++a) {
      const int m0 = m0s[a], j = j0s[a] + l16;
#pragma unroll
      for (int e = 0; e < 8; ++e) {
        const int m = m0 + e + 8 * half;
        const float4* qp = (const float4*)(qpre + m * 32);
        float4 q0 = qp[0], q1 = qp[1], q2 = qp[2], q3 = qp[3];
        float4 q4 = qp[4], q5 = qp[5], q6 = qp[6], q7 = qp[7];
        float fq = bqhv[a] + q0.x*wqv[a][0] + q0.y*wqv[a][1] + q0.z*wqv[a][2] + q0.w*wqv[a][3]
                           + q1.x*wqv[a][4] + q1.y*wqv[a][5] + q1.z*wqv[a][6] + q1.w*wqv[a][7];
        float iq = bqhv[a] + q2.x*wqv[a][0] + q2.y*wqv[a][1] + q2.z*wqv[a][2] + q2.w*wqv[a][3]
                           + q3.x*wqv[a][4] + q3.y*wqv[a][5] + q3.z*wqv[a][6] + q3.w*wqv[a][7];
        float gq = bqhv[a] + q4.x*wqv[a][0] + q4.y*wqv[a][1] + q4.z*wqv[a][2] + q4.w*wqv[a][3]
                           + q5.x*wqv[a][4] + q5.y*wqv[a][5] + q5.z*wqv[a][6] + q5.w*wqv[a][7];
        float oq = bqhv[a] + q6.x*wqv[a][0] + q6.y*wqv[a][1] + q6.z*wqv[a][2] + q6.w*wqv[a][3]
                           + q7.x*wqv[a][4] + q7.y*wqv[a][5] + q7.z*wqv[a][6] + q7.w*wqv[a][7];
        float F = mixv * sigm(af[a][e])  + (1.f - mixv) * fq;
        float I = mixv * sigm(ai_[a][e]) + (1.f - mixv) * iq;
        float G = mixv * tanhf(ag[a][e]) + (1.f - mixv) * gq;
        float O = mixv * sigm(ao[a][e])  + (1.f - mixv) * oq;
        float c = F * cxv[a][e] + I * G;
        cxv[a][e] = c;
        float hv = O * tanhf(c);
        size_t oidx = (size_t)t * 16384 + (size_t)m * 256 + j;
        outs[oidx]  = hv;
        outsH[oidx] = (_Float16)hv;
        comb[m * COMBLD + 128 + j] = (_Float16)hv;
        if (t == 1023) hxo[m * 256 + j] = hv;
      }
    }
    __syncthreads();  // hx/comb writes before next step
  }

#pragma unroll
  for (int a = 0; a < 2; ++a) {
    const int m0 = m0s[a], j = j0s[a] + l16;
#pragma unroll
    for (int e = 0; e < 8; ++e) cxo[(m0 + e + 8 * half) * 256 + j] = cxv[a][e];
  }
}

// ---------------- generic M x 256 GEMM with f16 A staged via async LDS ------
// MODE 0: outH[m*Ntot+n] = f16(acc + bias[n])           (qkv projection)
// MODE 1: outF[m*256+n] += mix * (acc + bias[n])        (out proj + residual)
template <int MODE>
__global__ __launch_bounds__(256, 2) void gemm_kernel(
    const _Float16* __restrict__ A,    // M x 256 f16
    const _Float16* __restrict__ W,    // Ntot x 256 f16
    const float* __restrict__ bias,    // Ntot
    _Float16* __restrict__ outH, float* __restrict__ outF, int Ntot,
    const float* __restrict__ mixp)
{
  __shared__ _Float16 Alds[64 * 264];
  const int tid = threadIdx.x, lane = tid & 31, wid = tid >> 5;
  const int l16 = lane & 15, half = lane >> 4, h8o = half * 8;
  const int bm = blockIdx.x;         // 64-row M tile
  const int n0 = blockIdx.y * 64;    // 64-col N tile
  const _Float16* Ag = A + (size_t)bm * 64 * 256;

  // stage the full 64x256 f16 A tile (32KB) asynchronously into LDS
#pragma unroll
  for (int i = 0; i < 8; ++i) {
    int ch = tid + i * 256;          // 2048 x 16B chunks
    int r = ch >> 5, c = (ch & 31) * 8;
    cp_async16(Alds + r * 264 + c, Ag + (size_t)r * 256 + c);
  }
  async_wait0();
  __syncthreads();

  const int ms = (wid & 3) * 16;     // m strip within tile
  const int nn = (wid >> 2) * 32;    // n half within tile
  v8f acc0 = splat8(0.f), acc1 = splat8(0.f);
#pragma unroll
  for (int ks = 0; ks < 8; ++ks) {
    int k0 = ks * 32;
    v16h a  = frag_ld(Alds + (ms + l16) * 264 + k0 + h8o);
    v16h b0 = frag_ld(W + (size_t)(n0 + nn + l16) * 256 + k0 + h8o);
    acc0 = wmma16(a, b0, acc0);
    v16h b1 = frag_ld(W + (size_t)(n0 + nn + 16 + l16) * 256 + k0 + h8o);
    acc1 = wmma16(a, b1, acc1);
  }
  const float mixv = (MODE == 1) ? mixp[0] : 0.f;
#pragma unroll
  for (int e = 0; e < 8; ++e) {
    int m = bm * 64 + ms + e + 8 * half;
    int na = n0 + nn + l16, nb = n0 + nn + 16 + l16;
    float va = acc0[e] + bias[na];
    float vb = acc1[e] + bias[nb];
    if (MODE == 0) {
      outH[(size_t)m * Ntot + na] = (_Float16)va;
      outH[(size_t)m * Ntot + nb] = (_Float16)vb;
    } else {
      size_t ia = (size_t)m * 256 + na, ib = (size_t)m * 256 + nb;
      outF[ia] = outF[ia] + mixv * va;
      outF[ib] = outF[ib] + mixv * vb;
    }
  }
}

// ---------------- attention: one block per (s, head), B=64 keys -------------
__global__ __launch_bounds__(256, 2) void attn_kernel(
    const _Float16* __restrict__ qkv,  // (S*B) x 768 f16
    _Float16* __restrict__ ctx)        // (S*B) x 256 f16
{
  __shared__ _Float16 qs[64 * 80];
  __shared__ _Float16 ks_[64 * 80];
  __shared__ _Float16 vT[64 * 80];     // transposed: [d][n]
  __shared__ float    sc[64 * 68];
  __shared__ _Float16 at[64 * 80];
  const int tid = threadIdx.x, lane = tid & 31, wid = tid >> 5;
  const int l16 = lane & 15, half = lane >> 4, h8o = half * 8;
  const int s = blockIdx.x >> 2, h = blockIdx.x & 3;
  const _Float16* base = qkv + (size_t)s * 64 * 768 + h * 64;

  // async-stage q and k tiles; transpose-stage v
#pragma unroll
  for (int i = 0; i < 2; ++i) {
    int ch = tid + i * 256;            // 512 x 16B chunks per matrix
    int r = ch >> 3, c = (ch & 7) * 8;
    cp_async16(qs  + r * 80 + c, base + (size_t)r * 768 + c);
    cp_async16(ks_ + r * 80 + c, base + 256 + (size_t)r * 768 + c);
    h8 v = *(const h8*)(base + 512 + (size_t)r * 768 + c);
#pragma unroll
    for (int u = 0; u < 8; ++u) vT[(c + u) * 80 + r] = v[u];
  }
  async_wait0();
  __syncthreads();

  // scores = q @ k^T * 1/8   (16 tiles of 16x16, 2 per wave)
#pragma unroll
  for (int a = 0; a < 2; ++a) {
    int tt = wid * 2 + a;
    int m0 = (tt >> 2) * 16, n0 = (tt & 3) * 16;
    v8f acc = splat8(0.f);
#pragma unroll
    for (int kk = 0; kk < 2; ++kk) {
      int k0 = kk * 32;
      v16h A = frag_ld(qs  + (m0 + l16) * 80 + k0 + h8o);
      v16h B = frag_ld(ks_ + (n0 + l16) * 80 + k0 + h8o);
      acc = wmma16(A, B, acc);
    }
#pragma unroll
    for (int e = 0; e < 8; ++e)
      sc[(m0 + e + 8 * half) * 68 + n0 + l16] = acc[e] * 0.125f;
  }
  __syncthreads();

  // row softmax (threads 0..63, one row each)
  if (tid < 64) {
    float mx = -1e30f;
    for (int n = 0; n < 64; ++n) mx = fmaxf(mx, sc[tid * 68 + n]);
    float sum = 0.f;
    for (int n = 0; n < 64; ++n) sum += __expf(sc[tid * 68 + n] - mx);
    float inv = 1.f / sum;
    for (int n = 0; n < 64; ++n)
      at[tid * 80 + n] = (_Float16)(__expf(sc[tid * 68 + n] - mx) * inv);
  }
  __syncthreads();

  // ctx = attn @ v   (B fragment from vT: rows = d, cols = n)
#pragma unroll
  for (int a = 0; a < 2; ++a) {
    int tt = wid * 2 + a;
    int m0 = (tt >> 2) * 16, d0 = (tt & 3) * 16;
    v8f acc = splat8(0.f);
#pragma unroll
    for (int kk = 0; kk < 2; ++kk) {
      int k0 = kk * 32;
      v16h A = frag_ld(at + (m0 + l16) * 80 + k0 + h8o);
      v16h B = frag_ld(vT + (d0 + l16) * 80 + k0 + h8o);
      acc = wmma16(A, B, acc);
    }
#pragma unroll
    for (int e = 0; e < 8; ++e) {
      int m = m0 + e + 8 * half, d = d0 + l16;
      ctx[((size_t)(s * 64 + m)) * 256 + h * 64 + d] = (_Float16)acc[e];
    }
  }
}

// ---------------- host-side launch ----------------
extern "C" void kernel_launch(void* const* d_in, const int* in_sizes, int n_in,
                              void* d_out, int out_size, void* d_ws, size_t ws_size,
                              hipStream_t stream) {
  (void)in_sizes; (void)n_in; (void)out_size; (void)ws_size;
  const float* x    = (const float*)d_in[0];
  const float* Wf   = (const float*)d_in[1];  const float* bf  = (const float*)d_in[2];
  const float* Wi   = (const float*)d_in[3];  const float* bi  = (const float*)d_in[4];
  const float* Wg   = (const float*)d_in[5];  const float* bg  = (const float*)d_in[6];
  const float* Wo   = (const float*)d_in[7];  const float* bo  = (const float*)d_in[8];
  const float* Wfq  = (const float*)d_in[9];  const float* bfq = (const float*)d_in[10];
  const float* Wiq  = (const float*)d_in[11]; const float* biq = (const float*)d_in[12];
  const float* Wgq  = (const float*)d_in[13]; const float* bgq = (const float*)d_in[14];
  const float* Woq  = (const float*)d_in[15]; const float* boq = (const float*)d_in[16];
  const float* Wqh  = (const float*)d_in[17]; const float* bqh = (const float*)d_in[18];
  const float* ipw  = (const float*)d_in[19]; const float* ipb = (const float*)d_in[20];
  const float* ow   = (const float*)d_in[21]; const float* ob  = (const float*)d_in[22];
  const float* mix  = (const float*)d_in[23];

  // workspace carve-up (all offsets 16B aligned)
  char* w = (char*)d_ws;
  _Float16* Wc4  = (_Float16*)w; w += (size_t)4 * 256 * 384 * 2;
  _Float16* Wq   = (_Float16*)w; w += (size_t)32 * 384 * 2;
  _Float16* Wip  = (_Float16*)w; w += (size_t)768 * 256 * 2;
  _Float16* Wout = (_Float16*)w; w += (size_t)256 * 256 * 2;
  _Float16* outH = (_Float16*)w; w += (size_t)1024 * 64 * 256 * 2;
  _Float16* qkvH = (_Float16*)w; w += (size_t)1024 * 64 * 768 * 2;
  _Float16* ctxH = (_Float16*)w; w += (size_t)1024 * 64 * 256 * 2;

  float* outs = (float*)d_out;
  float* hxo  = outs + (size_t)1024 * 64 * 256;
  float* cxo  = hxo + 64 * 256;

  auto cvt = [&](const float* s, _Float16* d, int n) {
    cvt_f32_to_f16<<<(n + 255) / 256, 256, 0, stream>>>(s, d, n);
  };
  cvt(Wf, Wc4 + (size_t)0 * 98304, 98304);
  cvt(Wi, Wc4 + (size_t)1 * 98304, 98304);
  cvt(Wg, Wc4 + (size_t)2 * 98304, 98304);
  cvt(Wo, Wc4 + (size_t)3 * 98304, 98304);
  cvt(Wfq, Wq + 0 * 3072, 3072);
  cvt(Wiq, Wq + 1 * 3072, 3072);
  cvt(Wgq, Wq + 2 * 3072, 3072);
  cvt(Woq, Wq + 3 * 3072, 3072);
  cvt(ipw, Wip, 768 * 256);
  cvt(ow, Wout, 256 * 256);

  scan_kernel<<<1, 1024, 0, stream>>>(x, Wc4, Wq, bf, bi, bg, bo,
                                      bfq, biq, bgq, boq, Wqh, bqh, mix,
                                      outs, outH, hxo, cxo);

  gemm_kernel<0><<<dim3(1024, 12), 256, 0, stream>>>(outH, Wip, ipb, qkvH,
                                                     nullptr, 768, nullptr);
  attn_kernel<<<4096, 256, 0, stream>>>(qkvH, ctxH);
  gemm_kernel<1><<<dim3(1024, 4), 256, 0, stream>>>(ctxH, Wout, ob, nullptr,
                                                    outs, 256, mix);
}
